// EncoderBlock_19000935318051
// MI455X (gfx1250) — matrix operand order repaired and checked by reading, not compiled, not run on hardware
//
#include <hip/hip_runtime.h>
#include <hip/hip_bf16.h>

// ---------------------------------------------------------------------------
// Transformer encoder block for MI455X (gfx1250), bf16 WMMA + f32 accumulate.
// GEMM tiles are streamed global->LDS with CDNA5 async-to-LDS instructions
// (ASYNCcnt) and double-buffered so WMMAs overlap the next tile's DMA.
// ---------------------------------------------------------------------------

typedef __attribute__((ext_vector_type(16))) __bf16 v16bf;
typedef __attribute__((ext_vector_type(8)))  float  v8f;

union FragU { v16bf v; uint4 u[2]; };

// A-matrix 16x32 bf16 fragment (ISA 7.12.2): lane l<16 holds row l,
// K {0..7, 16..23}; lane l+16 holds row l, K {8..15, 24..31}.
__device__ __forceinline__ v16bf ldsA16x32(const __bf16* base, int stride) {
  const int lane = threadIdx.x & 31;
  const int row  = lane & 15;
  const int kb   = (lane >> 4) * 8;
  const __bf16* p = base + row * stride;
  FragU f;
  f.u[0] = *(const uint4*)(p + kb);
  f.u[1] = *(const uint4*)(p + kb + 16);
  return f.v;
}

// B-matrix 32x16 bf16 fragment: lane l holds column (l&15),
// K {0..7, 16..23} (l<16) or {8..15, 24..31}. Source tile is (n, k) row-major in LDS.
__device__ __forceinline__ v16bf ldsB32x16(const __bf16* base, int stride) {
  const int lane = threadIdx.x & 31;
  const int n  = lane & 15;
  const int kb = (lane >> 4) * 8;
  const __bf16* p = base + n * stride + kb;
  FragU f;
  f.u[0] = *(const uint4*)(p);
  f.u[1] = *(const uint4*)(p + 16);
  return f.v;
}

// CDNA5 async copy: 16B global -> LDS, no VGPR staging, tracked by ASYNCcnt.
__device__ __forceinline__ void async_b128(__bf16* lds, const __bf16* g) {
  const unsigned loff = (unsigned)(uintptr_t)lds;  // low 32 bits = LDS offset
  asm volatile("global_load_async_to_lds_b128 %0, %1, off"
               :: "v"(loff), "v"(g) : "memory");
}

__device__ __forceinline__ void wait_async0() {
  asm volatile("s_wait_asynccnt 0x0" ::: "memory");
}

// ---------------------------------------------------------------------------
// fp32 -> bf16 elementwise convert (weights)
// ---------------------------------------------------------------------------
__global__ void k_cvt_bf16(const float* __restrict__ src,
                           __bf16* __restrict__ dst, int n) {
  for (int i = blockIdx.x * blockDim.x + threadIdx.x; i < n;
       i += gridDim.x * blockDim.x)
    dst[i] = (__bf16)src[i];
}

__global__ void k_pack_qkv_bias(const float* __restrict__ bq,
                                const float* __restrict__ bk,
                                const float* __restrict__ bv,
                                float* __restrict__ out) {
  const int i = blockIdx.x * 256 + threadIdx.x;
  out[i] = (i < 1024) ? bq[i] : ((i < 2048) ? bk[i - 1024] : bv[i - 2048]);
}

// ---------------------------------------------------------------------------
// LayerNorm (torch-style: unbiased std ddof=1, scalar alpha/beta,
// divide by (std+eps)) ; fp32 in -> bf16 out. One block per row of 1024.
// ---------------------------------------------------------------------------
__global__ __launch_bounds__(256) void k_layernorm_bf16(
    const float* __restrict__ X, __bf16* __restrict__ Y,
    const float* __restrict__ alpha, const float* __restrict__ beta) {
  const int row = blockIdx.x;
  const int t = threadIdx.x;
  const float4 xv = *(const float4*)(X + (size_t)row * 1024 + t * 4);
  float s  = xv.x + xv.y + xv.z + xv.w;
  float ss = xv.x * xv.x + xv.y * xv.y + xv.z * xv.z + xv.w * xv.w;
#pragma unroll
  for (int d = 1; d < 32; d <<= 1) {
    s  += __shfl_xor(s, d, 32);
    ss += __shfl_xor(ss, d, 32);
  }
  __shared__ float rs[8], rss[8];
  if ((t & 31) == 0) { rs[t >> 5] = s; rss[t >> 5] = ss; }
  __syncthreads();
  float tot = 0.f, tot2 = 0.f;
#pragma unroll
  for (int i = 0; i < 8; ++i) { tot += rs[i]; tot2 += rss[i]; }
  const float mean = tot * (1.0f / 1024.0f);
  const float var  = fmaxf(tot2 - 1024.0f * mean * mean, 0.0f) * (1.0f / 1023.0f);
  const float sc   = alpha[0] / (sqrtf(var) + 1e-6f);
  const float sh   = beta[0];
  __bf16* y = Y + (size_t)row * 1024 + t * 4;
  y[0] = (__bf16)((xv.x - mean) * sc + sh);
  y[1] = (__bf16)((xv.y - mean) * sc + sh);
  y[2] = (__bf16)((xv.z - mean) * sc + sh);
  y[3] = (__bf16)((xv.w - mean) * sc + sh);
}

// ---------------------------------------------------------------------------
// Tiled GEMM: C(M,N) = A(M,K)bf16 @ W(N,K)bf16^T  (+bias, +optional relu,
// +optional fp32 residual). Optional fp32 and/or bf16 outputs.
// Block tile 128x128, 8 waves (4x2), wave tile 32x64 = 2x4 WMMA 16x16x32.
// Tiles land in double-buffered LDS via global_load_async_to_lds_b128, so
// the next k-step's DMA overlaps this k-step's WMMAs; one barrier per k-step.
// LDS stride 40 bf16 (80 B rows) keeps 16B fragment alignment, staggers banks.
// ---------------------------------------------------------------------------
__global__ __launch_bounds__(256) void k_gemm_bf16(
    const __bf16* __restrict__ A, const __bf16* __restrict__ W,
    const float* __restrict__ bias, const float* __restrict__ resid,
    float* __restrict__ outF, __bf16* __restrict__ outB,
    int M, int N, int K, int relu) {
  __shared__ __align__(16) __bf16 sA[2][128 * 40];
  __shared__ __align__(16) __bf16 sB[2][128 * 40];
  const int tid = threadIdx.x;
  const int bm = blockIdx.y * 128;
  const int bn = blockIdx.x * 128;
  const int wave = tid >> 5;
  const int wm = (wave & 3) * 32;   // 4 waves along M
  const int wn = (wave >> 2) * 64;  // 2 waves along N

  // each thread DMAs 2 A-chunks + 2 B-chunks (16B each) per k-step
  const int ldr0 = tid >> 2, ldq = (tid & 3) * 8;   // rows 0..63
  const int ldr1 = ldr0 + 64;                       // rows 64..127

  auto issue_tiles = [&](int buf, int k0) {
    const __bf16* Ag = A + (size_t)bm * K + k0;
    const __bf16* Bg = W + (size_t)bn * K + k0;
    async_b128(&sA[buf][ldr0 * 40 + ldq], Ag + (size_t)ldr0 * K + ldq);
    async_b128(&sA[buf][ldr1 * 40 + ldq], Ag + (size_t)ldr1 * K + ldq);
    async_b128(&sB[buf][ldr0 * 40 + ldq], Bg + (size_t)ldr0 * K + ldq);
    async_b128(&sB[buf][ldr1 * 40 + ldq], Bg + (size_t)ldr1 * K + ldq);
  };

  v8f c[2][4] = {};
  issue_tiles(0, 0);
  int p = 0;
  for (int k0 = 0; k0 < K; k0 += 32) {
    wait_async0();       // our async writes into buffer p have landed
    __syncthreads();     // everyone's have; prior reads of p^1 are done
    if (k0 + 32 < K) issue_tiles(p ^ 1, k0 + 32);

    const __bf16* sAb = &sA[p][0];
    const __bf16* sBb = &sB[p][0];
    const v16bf a0 = ldsA16x32(sAb + (wm + 0)  * 40, 40);
    const v16bf a1 = ldsA16x32(sAb + (wm + 16) * 40, 40);
    v16bf bf0 = ldsB32x16(sBb + (wn + 0)  * 40, 40);
    v16bf bf1 = ldsB32x16(sBb + (wn + 16) * 40, 40);
    v16bf bf2 = ldsB32x16(sBb + (wn + 32) * 40, 40);
    v16bf bf3 = ldsB32x16(sBb + (wn + 48) * 40, 40);
    c[0][0] = __builtin_amdgcn_wmma_f32_16x16x32_bf16(false, a0, false, bf0, (short)0, c[0][0], false, false);
    c[1][0] = __builtin_amdgcn_wmma_f32_16x16x32_bf16(false, a1, false, bf0, (short)0, c[1][0], false, false);
    c[0][1] = __builtin_amdgcn_wmma_f32_16x16x32_bf16(false, a0, false, bf1, (short)0, c[0][1], false, false);
    c[1][1] = __builtin_amdgcn_wmma_f32_16x16x32_bf16(false, a1, false, bf1, (short)0, c[1][1], false, false);
    c[0][2] = __builtin_amdgcn_wmma_f32_16x16x32_bf16(false, a0, false, bf2, (short)0, c[0][2], false, false);
    c[1][2] = __builtin_amdgcn_wmma_f32_16x16x32_bf16(false, a1, false, bf2, (short)0, c[1][2], false, false);
    c[0][3] = __builtin_amdgcn_wmma_f32_16x16x32_bf16(false, a0, false, bf3, (short)0, c[0][3], false, false);
    c[1][3] = __builtin_amdgcn_wmma_f32_16x16x32_bf16(false, a1, false, bf3, (short)0, c[1][3], false, false);
    p ^= 1;
  }
  // C/D layout: VGPR r -> row r + (lane>=16)*8 ; col = lane&15
  const int lane = tid & 31;
  const int cn = lane & 15;
  const int rh = (lane >> 4) * 8;
#pragma unroll
  for (int fm = 0; fm < 2; ++fm)
#pragma unroll
    for (int fb = 0; fb < 4; ++fb)
#pragma unroll
      for (int r = 0; r < 8; ++r) {
        const int row = bm + wm + fm * 16 + rh + r;
        const int col = bn + wn + fb * 16 + cn;
        float v = c[fm][fb][r] + bias[col];
        if (relu) v = fmaxf(v, 0.f);
        if (resid) v += resid[(size_t)row * N + col];
        if (outF) outF[(size_t)row * N + col] = v;
        if (outB) outB[(size_t)row * N + col] = (__bf16)v;
      }
}

// ---------------------------------------------------------------------------
// Flash attention: qkv packed (B*S, 3*1024) bf16 -> ctx (B*S, 1024) bf16.
// Grid (S/64, B*H). 4 waves, each owns 16 query rows; streams K-blocks of 32.
// Q fragments held in registers; K tile row-major LDS; V tile transposed in
// LDS so PV B-fragments are contiguous; P re-laid-out C->A via per-wave LDS.
// ---------------------------------------------------------------------------
__global__ __launch_bounds__(128) void k_flash_attn(
    const __bf16* __restrict__ qkv, const int* __restrict__ mask,
    __bf16* __restrict__ ctx) {
  __shared__ __align__(16) __bf16 sK[32 * 72];      // (key, d)
  __shared__ __align__(16) __bf16 sVt[64 * 40];     // (d, key)
  __shared__ __align__(16) __bf16 sP[4][16 * 40];   // per-wave P staging
  const int bh = blockIdx.y;
  const int b = bh >> 4;
  const int h = bh & 15;
  const int tid = threadIdx.x;
  const int wave = tid >> 5;
  const int lane = tid & 31;
  const int qrow0 = blockIdx.x * 64 + wave * 16;
  const size_t rstride = 3072;
  const __bf16* Qb = qkv + (size_t)b * 1024 * rstride + h * 64;
  const __bf16* Kb = Qb + 1024;
  const __bf16* Vb = Qb + 2048;

  // Q fragments (A layout) straight from global: d in [0,32) and [32,64)
  const int ar  = lane & 15;
  const int akb = (lane >> 4) * 8;
  const __bf16* qp = Qb + (size_t)(qrow0 + ar) * rstride;
  FragU qa0, qa1;
  qa0.u[0] = *(const uint4*)(qp + akb);
  qa0.u[1] = *(const uint4*)(qp + akb + 16);
  qa1.u[0] = *(const uint4*)(qp + 32 + akb);
  qa1.u[1] = *(const uint4*)(qp + 32 + akb + 16);

  float rMax[8], rSum[8];
#pragma unroll
  for (int r = 0; r < 8; ++r) { rMax[r] = -3.0e38f; rSum[r] = 0.f; }
  v8f o[4] = {};

  const int cn = lane & 15;
  const int rh = (lane >> 4) * 8;
  const int* mrow = mask + b * 1024;   // mask shape (8,1,1,1024)

  for (int j = 0; j < 1024; j += 32) {
#pragma unroll
    for (int it = 0; it < 2; ++it) {             // K tile: 32 rows x 8 uint4
      const int idx = tid + it * 128;
      const int kr = idx >> 3, q = idx & 7;
      *(uint4*)(sK + kr * 72 + q * 8) =
          *(const uint4*)(Kb + (size_t)(j + kr) * rstride + q * 8);
    }
#pragma unroll
    for (int it = 0; it < 2; ++it) {             // V tile, transposed scatter
      const int idx = tid + it * 128;
      const int vr = idx >> 3, q = idx & 7;
      const uint4 g = *(const uint4*)(Vb + (size_t)(j + vr) * rstride + q * 8);
      const __bf16* ge = (const __bf16*)&g;
#pragma unroll
      for (int e = 0; e < 8; ++e)
        sVt[(q * 8 + e) * 40 + vr] = ge[e];
    }
    __syncthreads();

    // scores: S(16x32) = Q(16x64) @ Kblk^T(64x32), two 16-col fragments
    v8f s[2];
#pragma unroll
    for (int fn = 0; fn < 2; ++fn) {
      const v16bf bk0 = ldsB32x16(sK + fn * 16 * 72, 72);        // d 0..31
      const v16bf bk1 = ldsB32x16(sK + fn * 16 * 72 + 32, 72);   // d 32..63
      v8f z = {};
      z = __builtin_amdgcn_wmma_f32_16x16x32_bf16(false, qa0.v, false, bk0, (short)0, z, false, false);
      z = __builtin_amdgcn_wmma_f32_16x16x32_bf16(false, qa1.v, false, bk1, (short)0, z, false, false);
      s[fn] = z;
    }
    // scale 1/sqrt(64) + mask
#pragma unroll
    for (int fn = 0; fn < 2; ++fn) {
      const float mv = (mrow[j + fn * 16 + cn] == 0) ? -1.0e9f : 0.0f;
#pragma unroll
      for (int r = 0; r < 8; ++r)
        s[fn][r] = s[fn][r] * 0.125f + mv;
    }
    // online softmax: each row lives across the 16 lanes of a half-wave
#pragma unroll
    for (int r = 0; r < 8; ++r) {
      float mx = fmaxf(s[0][r], s[1][r]);
      mx = fmaxf(mx, __shfl_xor(mx, 1, 16));
      mx = fmaxf(mx, __shfl_xor(mx, 2, 16));
      mx = fmaxf(mx, __shfl_xor(mx, 4, 16));
      mx = fmaxf(mx, __shfl_xor(mx, 8, 16));
      const float mnew = fmaxf(rMax[r], mx);
      const float corr = __expf(rMax[r] - mnew);
      rMax[r] = mnew;
      const float p0 = __expf(s[0][r] - mnew);
      const float p1 = __expf(s[1][r] - mnew);
      s[0][r] = p0; s[1][r] = p1;
      float ps = p0 + p1;
      ps += __shfl_xor(ps, 1, 16);
      ps += __shfl_xor(ps, 2, 16);
      ps += __shfl_xor(ps, 4, 16);
      ps += __shfl_xor(ps, 8, 16);
      rSum[r] = rSum[r] * corr + ps;
#pragma unroll
      for (int fo = 0; fo < 4; ++fo) o[fo][r] *= corr;
    }
    // C-layout -> A-layout via per-wave LDS (same-wave DS ops are in-order)
    __bf16* pst = &sP[wave][0];
#pragma unroll
    for (int fn = 0; fn < 2; ++fn)
#pragma unroll
      for (int r = 0; r < 8; ++r)
        pst[(rh + r) * 40 + fn * 16 + cn] = (__bf16)s[fn][r];
    const v16bf ap = ldsA16x32(pst, 40);
    // O(16x64) += P(16x32) @ V(32x64)
#pragma unroll
    for (int fo = 0; fo < 4; ++fo) {
      const v16bf bv = ldsB32x16(sVt + fo * 16 * 40, 40);
      o[fo] = __builtin_amdgcn_wmma_f32_16x16x32_bf16(false, ap, false, bv, (short)0, o[fo], false, false);
    }
    __syncthreads();
  }
  // normalize + store ctx (b, s, h*64+d) bf16
#pragma unroll
  for (int fo = 0; fo < 4; ++fo)
#pragma unroll
    for (int r = 0; r < 8; ++r) {
      const int srow = qrow0 + rh + r;
      ctx[((size_t)(b * 1024 + srow)) * 1024 + h * 64 + fo * 16 + cn] =
          (__bf16)(o[fo][r] / rSum[r]);
    }
}

// ---------------------------------------------------------------------------
// Host orchestration
// ---------------------------------------------------------------------------
extern "C" void kernel_launch(void* const* d_in, const int* in_sizes, int n_in,
                              void* d_out, int out_size, void* d_ws, size_t ws_size,
                              hipStream_t stream) {
  (void)in_sizes; (void)n_in; (void)out_size; (void)ws_size;
  const float* X    = (const float*)d_in[0];
  const int*   mask = (const int*)  d_in[1];
  const float* wq   = (const float*)d_in[2];
  const float* bq   = (const float*)d_in[3];
  const float* wk   = (const float*)d_in[4];
  const float* bk   = (const float*)d_in[5];
  const float* wv   = (const float*)d_in[6];
  const float* bv   = (const float*)d_in[7];
  const float* wo   = (const float*)d_in[8];
  const float* bo   = (const float*)d_in[9];
  const float* w1   = (const float*)d_in[10];
  const float* b1   = (const float*)d_in[11];
  const float* w2   = (const float*)d_in[12];
  const float* b2   = (const float*)d_in[13];
  const float* ln1a = (const float*)d_in[14];
  const float* ln1b = (const float*)d_in[15];
  const float* ln2a = (const float*)d_in[16];
  const float* ln2b = (const float*)d_in[17];
  float* out = (float*)d_out;

  char* ws = (char*)d_ws;
  size_t off = 0;
  auto take = [&](size_t bytes) {
    void* p = ws + off;
    off += (bytes + 255) & ~(size_t)255;
    return p;
  };
  __bf16* wqkv = (__bf16*)take(3072ull * 1024 * 2);  // [wq;wk;wv] bf16 (N,K)
  __bf16* wob  = (__bf16*)take(1024ull * 1024 * 2);
  __bf16* w1b  = (__bf16*)take(4096ull * 1024 * 2);
  __bf16* w2b  = (__bf16*)take(1024ull * 4096 * 2);
  float*  bqkv = (float*) take(3072ull * 4);
  __bf16* xn   = (__bf16*)take(8192ull * 1024 * 2);  // ln output (reused)
  float*  x1   = (float*) take(8192ull * 1024 * 4);  // post-attn residual
  __bf16* qkv  = (__bf16*)take(8192ull * 3072 * 2);  // fused QKV activations
  __bf16* ctx  = (__bf16*)take(8192ull * 1024 * 2);  // attention output
  __bf16* hb   = qkv;  // FFN hidden (8192x4096 bf16) reuses qkv+ctx (exact fit)

  // 1) weights fp32 -> bf16
  k_cvt_bf16<<<1024, 256, 0, stream>>>(wq, wqkv,                 1024 * 1024);
  k_cvt_bf16<<<1024, 256, 0, stream>>>(wk, wqkv + 1024ull*1024,  1024 * 1024);
  k_cvt_bf16<<<1024, 256, 0, stream>>>(wv, wqkv + 2048ull*1024,  1024 * 1024);
  k_cvt_bf16<<<1024, 256, 0, stream>>>(wo, wob,                  1024 * 1024);
  k_cvt_bf16<<<2048, 256, 0, stream>>>(w1, w1b,                  4096 * 1024);
  k_cvt_bf16<<<2048, 256, 0, stream>>>(w2, w2b,                  1024 * 4096);
  k_pack_qkv_bias<<<12, 256, 0, stream>>>(bq, bk, bv, bqkv);

  // 2) LN1
  k_layernorm_bf16<<<8192, 256, 0, stream>>>(X, xn, ln1a, ln1b);
  // 3) fused QKV projection (M=8192, N=3072, K=1024)
  k_gemm_bf16<<<dim3(3072 / 128, 8192 / 128), 256, 0, stream>>>(
      xn, wqkv, bqkv, nullptr, nullptr, qkv, 8192, 3072, 1024, 0);
  // 4) flash attention -> ctx
  k_flash_attn<<<dim3(16, 128), 128, 0, stream>>>(qkv, mask, ctx);
  // 5) output projection + residual: x1 = X + ctx @ wo^T + bo
  k_gemm_bf16<<<dim3(1024 / 128, 8192 / 128), 256, 0, stream>>>(
      ctx, wob, bo, X, x1, nullptr, 8192, 1024, 1024, 0);
  // 6) LN2
  k_layernorm_bf16<<<8192, 256, 0, stream>>>(x1, xn, ln2a, ln2b);
  // 7) FFN1: h = relu(xn @ w1^T + b1) (bf16)
  k_gemm_bf16<<<dim3(4096 / 128, 8192 / 128), 256, 0, stream>>>(
      xn, w1b, b1, nullptr, nullptr, hb, 8192, 4096, 1024, 1);
  // 8) FFN2 + residual: out = x1 + h @ w2^T + b2
  k_gemm_bf16<<<dim3(1024 / 128, 8192 / 128), 256, 0, stream>>>(
      hb, w2b, b2, x1, out, nullptr, 8192, 1024, 4096, 0);
}
